// PINODEv3_13907104104482
// MI455X (gfx1250) — matrix operand order repaired and checked by reading, not compiled, not run on hardware
//
#include <hip/hip_runtime.h>

#define B_TOT   8192
#define T_STEPS 32
#define H       256
#define NLM1    3
#define M_TILE  64
#define HINS    132   // packed-bf16 row stride in words (16B aligned)
#define HFS     260   // fp32 row stride in floats

typedef __bf16 v16bf __attribute__((ext_vector_type(16)));
typedef float  v8f   __attribute__((ext_vector_type(8)));

union Frag {
    unsigned u[8];
    uint4    q[2];
    v16bf    v;
};

__device__ __forceinline__ unsigned f2bf(float f) {
    unsigned u = __builtin_bit_cast(unsigned, f);
    u += 0x7FFFu + ((u >> 16) & 1u);            // round-to-nearest-even
    return u >> 16;
}
__device__ __forceinline__ float bf2f(unsigned h) {
    return __builtin_bit_cast(float, h << 16);
}

// ---------------------------------------------------------------------------
// Prep: convert Wh (3 x 256 x 256 fp32) into bf16 pairs pre-swizzled into the
// per-lane WMMA B-fragment layout:
//   frag(l, kc, nt): lane L holds VGPR i = pair( W[32kc+16(L>>4)+2i][16nt+(L&15)] )
// Flat word index = (((l*8+kc)*16+nt)*32+lane)*8 + i  -> lane gets 32 contig bytes.
// ---------------------------------------------------------------------------
__global__ void prep_weights(const float* __restrict__ Wh,
                             unsigned* __restrict__ wsW) {
    int idx = blockIdx.x * blockDim.x + threadIdx.x;
    if (idx >= 3 * 8 * 16 * 32 * 8) return;
    int i    = idx & 7;
    int lane = (idx >> 3) & 31;
    int nt   = (idx >> 8) & 15;
    int kc   = (idx >> 12) & 7;
    int l    = idx >> 15;
    int k = 32 * kc + 16 * (lane >> 4) + 2 * i;
    int n = 16 * nt + (lane & 15);
    const float* Wl = Wh + l * H * H;
    unsigned lo = f2bf(Wl[k * H + n]);
    unsigned hi = f2bf(Wl[(k + 1) * H + n]);
    wsW[idx] = lo | (hi << 16);
}

// ---------------------------------------------------------------------------
// LayerNorm + SiLU over hF (fp32 [64][HFS]) -> packed bf16 hIn ([64][HINS])
// Wave w handles rows w*8 .. w*8+7; lane holds 8 contiguous columns.
// ---------------------------------------------------------------------------
__device__ __forceinline__ void ln_silu_pack(int li, float* hF, unsigned* hIn,
                                             const float* __restrict__ ln_g,
                                             const float* __restrict__ ln_b) {
    int lane = threadIdx.x & 31;
    int w    = threadIdx.x >> 5;
    const float* g = ln_g + li * H;
    const float* b = ln_b + li * H;
    for (int q = 0; q < 8; ++q) {
        int row = w * 8 + q;
        float v[8], s = 0.f, ss = 0.f;
#pragma unroll
        for (int j = 0; j < 8; ++j) {
            v[j] = hF[row * HFS + lane * 8 + j];
            s += v[j];
            ss += v[j] * v[j];
        }
#pragma unroll
        for (int off = 16; off > 0; off >>= 1) {
            s  += __shfl_xor(s, off, 32);
            ss += __shfl_xor(ss, off, 32);
        }
        float mu   = s * (1.f / 256.f);
        float var  = ss * (1.f / 256.f) - mu * mu;
        float rstd = rsqrtf(var + 1e-5f);
        uint4 pk;
        unsigned pkw[4];
#pragma unroll
        for (int j2 = 0; j2 < 4; ++j2) {
            int   n  = lane * 8 + 2 * j2;
            float x0 = (v[2 * j2]     - mu) * rstd * g[n]     + b[n];
            float x1 = (v[2 * j2 + 1] - mu) * rstd * g[n + 1] + b[n + 1];
            float a0 = __fdividef(x0, 1.f + __expf(-x0));
            float a1 = __fdividef(x1, 1.f + __expf(-x1));
            pkw[j2] = f2bf(a0) | (f2bf(a1) << 16);
        }
        pk.x = pkw[0]; pk.y = pkw[1]; pk.z = pkw[2]; pk.w = pkw[3];
        *(uint4*)&hIn[row * HINS + lane * 4] = pk;   // one ds_store_b128
    }
}

// ---------------------------------------------------------------------------
// Hidden GEMM: hF[64][256] = dequant(hIn) @ Wh[layer] + bh[layer]
// Wave w owns N-tiles {2w, 2w+1} across ALL 4 row tiles -> every weight
// fragment is loaded exactly once per workgroup (min L2 traffic), and the
// 8 WMMAs per k-chunk hit 8 distinct accumulators (no RAW chain).
// ---------------------------------------------------------------------------
__device__ __forceinline__ void gemm_hidden(int layer, const uint4* __restrict__ wq,
                                            const unsigned* hIn, float* hF,
                                            const float* __restrict__ bh) {
    int lane = threadIdx.x & 31;
    int w    = threadIdx.x >> 5;
    int nt0  = 2 * w;
    int half = lane >> 4;
    int mcol = lane & 15;

    v8f acc[4][2];
#pragma unroll
    for (int r = 0; r < 4; ++r)
#pragma unroll
        for (int p = 0; p < 2; ++p)
            acc[r][p] = {};

#pragma unroll
    for (int kc = 0; kc < 8; ++kc) {
        // A fragments for all 4 row tiles: 2 aligned b128 LDS loads each.
        Frag afr[4];
#pragma unroll
        for (int r = 0; r < 4; ++r) {
            int base = (16 * r + mcol) * HINS + 16 * kc + 8 * half;
            afr[r].q[0] = *(const uint4*)&hIn[base];
            afr[r].q[1] = *(const uint4*)&hIn[base + 4];
        }
#pragma unroll
        for (int p = 0; p < 2; ++p) {
            int nt = nt0 + p;
            const uint4* ptr =
                wq + ((((((layer << 3) | kc) << 4) | nt) << 5 | lane) << 1);
            if (kc < 7) {
                const uint4* np =
                    wq + ((((((layer << 3) | (kc + 1)) << 4) | nt) << 5 | lane) << 1);
                __builtin_prefetch((const void*)np, 0, 1);
            }
            Frag bf;
            bf.q[0] = ptr[0];
            bf.q[1] = ptr[1];
#pragma unroll
            for (int r = 0; r < 4; ++r)
                acc[r][p] = __builtin_amdgcn_wmma_f32_16x16x32_bf16(
                    false, afr[r].v, false, bf.v, (short)0, acc[r][p], false, false);
        }
    }

#pragma unroll
    for (int p = 0; p < 2; ++p) {
        int   n    = 16 * (nt0 + p) + mcol;
        float bias = bh[layer * H + n];
#pragma unroll
        for (int r = 0; r < 4; ++r)
#pragma unroll
            for (int i = 0; i < 8; ++i)
                hF[(16 * r + i + 8 * half) * HFS + n] = acc[r][p][i] + bias;
    }
}

// ---------------------------------------------------------------------------
// Main: each workgroup owns 64 batch rows for all 31 RK4 steps.
// ---------------------------------------------------------------------------
__global__ __launch_bounds__(256)
void pinode_main(const float* __restrict__ y0, const float* __restrict__ ts,
                 const float* __restrict__ params, const float* __restrict__ W0,
                 const float* __restrict__ b0, const float* __restrict__ bh,
                 const float* __restrict__ ln_g, const float* __restrict__ ln_b,
                 const float* __restrict__ Wout, const float* __restrict__ bout,
                 const float* __restrict__ gatep, const uint4* __restrict__ wq,
                 float* __restrict__ out) {
    extern __shared__ __align__(16) char smem_raw[];
    float*    hF   = (float*)smem_raw;              // 64*HFS fp32
    unsigned* hIn  = (unsigned*)(hF + 64 * HFS);    // 64*HINS packed bf16 pairs
    float*    xin  = (float*)(hIn + 64 * HINS);     // 64*6
    float*    yS   = xin + 64 * 6;                  // 192
    float*    yev  = yS + 192;                      // 192
    float*    kcur = yev + 192;                     // 192
    float*    kacc = kcur + 192;                    // 192

    int   tid  = threadIdx.x;
    int   row0 = blockIdx.x * M_TILE;
    float gate = gatep[0];

    if (tid < 192) {
        float y   = y0[row0 * 3 + tid];
        yS[tid]   = y;
        kcur[tid] = 0.f;
        out[row0 * 3 + tid] = y;                    // t = 0 slice
    }
    __syncthreads();

    for (int step = 0; step < T_STEPS - 1; ++step) {
        float dt = ts[step + 1] - ts[step];
        if (tid < 192) kacc[tid] = 0.f;

        for (int e = 0; e < 4; ++e) {
            float coef = (e == 0) ? 0.f : (e == 3) ? dt : 0.5f * dt;
            float kwgt = (e == 0 || e == 3) ? 1.f : 2.f;
            __syncthreads();
            if (tid < 192) yev[tid] = yS[tid] + coef * kcur[tid];
            __syncthreads();

            for (int idx = tid; idx < 64 * 6; idx += 256) {
                int m = idx / 6, j = idx % 6;
                xin[idx] = (j < 3) ? yev[m * 3 + j]
                                   : params[(row0 + m) * 3 + (j - 3)];
            }
            __syncthreads();

            // input layer (6 -> 256): thread tid = output column
            {
                int   n = tid;
                float w0c[6];
#pragma unroll
                for (int j = 0; j < 6; ++j) w0c[j] = W0[j * H + n];
                float bias = b0[n];
                for (int m = 0; m < 64; ++m) {
                    float a = bias;
#pragma unroll
                    for (int j = 0; j < 6; ++j) a += xin[m * 6 + j] * w0c[j];
                    hF[m * HFS + n] = a;
                }
            }
            __syncthreads();
            ln_silu_pack(0, hF, hIn, ln_g, ln_b);
            __syncthreads();

            for (int l = 0; l < NLM1; ++l) {
                gemm_hidden(l, wq, hIn, hF, bh);
                __syncthreads();
                ln_silu_pack(l + 1, hF, hIn, ln_g, ln_b);
                __syncthreads();
            }

            // output layer (256 -> 3) + Bloch RHS + RK4 slope update
            if (tid < 192) {
                int   m = tid / 3, c = tid % 3;
                float sacc = 0.f;
                for (int k2 = 0; k2 < 128; ++k2) {
                    unsigned uu = hIn[m * HINS + k2];
                    sacc += bf2f(uu & 0xFFFFu) * Wout[(2 * k2) * 3 + c]
                          + bf2f(uu >> 16)     * Wout[(2 * k2 + 1) * 3 + c];
                }
                float nv = sacc + bout[c];
                float u = yev[m * 3 + 0], v = yev[m * 3 + 1], ww = yev[m * 3 + 2];
                float Om = params[(row0 + m) * 3 + 0];
                float ga = params[(row0 + m) * 3 + 2];
                float bl = (c == 0) ? (-ga * u)
                         : (c == 1) ? (-ga * v - Om * ww)
                                    : (-2.f * ga * (ww + 1.f) + Om * v);
                float kv  = bl + gate * nv;
                kcur[tid] = kv;
                kacc[tid] += kwgt * kv;
            }
        }
        __syncthreads();
        if (tid < 192) {
            float yn = yS[tid] + (dt * (1.f / 6.f)) * kacc[tid];
            yS[tid]  = yn;
            out[(size_t)(step + 1) * B_TOT * 3 + row0 * 3 + tid] = yn;
        }
        __syncthreads();
    }
}

extern "C" void kernel_launch(void* const* d_in, const int* in_sizes, int n_in,
                              void* d_out, int out_size, void* d_ws, size_t ws_size,
                              hipStream_t stream) {
    const float* y0     = (const float*)d_in[0];
    const float* ts     = (const float*)d_in[1];
    const float* params = (const float*)d_in[2];
    const float* W0     = (const float*)d_in[3];
    const float* b0     = (const float*)d_in[4];
    const float* Wh     = (const float*)d_in[5];
    const float* bh     = (const float*)d_in[6];
    const float* ln_g   = (const float*)d_in[7];
    const float* ln_b   = (const float*)d_in[8];
    const float* Wout   = (const float*)d_in[9];
    const float* bout   = (const float*)d_in[10];
    const float* gate   = (const float*)d_in[11];
    float*    out = (float*)d_out;
    unsigned* wsW = (unsigned*)d_ws;     // 3*8*16*32*8 words = 384 KB

    prep_weights<<<384, 256, 0, stream>>>(Wh, wsW);

    const size_t smem =
        (size_t)(64 * HFS + 64 * HINS + 64 * 6 + 4 * 192) * sizeof(float);
    (void)hipFuncSetAttribute((const void*)pinode_main,
                              hipFuncAttributeMaxDynamicSharedMemorySize,
                              (int)smem);
    pinode_main<<<B_TOT / M_TILE, 256, smem, stream>>>(
        y0, ts, params, W0, b0, bh, ln_g, ln_b, Wout, bout, gate,
        (const uint4*)wsW, out);
}
